// NSA_63359357551390
// MI455X (gfx1250) — matrix operand order repaired and checked by reading, not compiled, not run on hardware
//
#include <hip/hip_runtime.h>
#include <math.h>

#define T_    2048
#define DIM_  2048
#define H_    16
#define KV_   4
#define D_    128
#define REP_  4
#define L_    32
#define S_    16
#define LP_   64
#define WIN_  512
#define TC_   127
#define NS_   32
#define QKVD_ 3072
#define KVD_  512
#define HD_   2048
#define CMPH_ 256
#define GH_   512
#define SCALE_ 0.08838834764831845f

typedef __attribute__((ext_vector_type(16))) __bf16 v16bf;
typedef __attribute__((ext_vector_type(8)))  float  v8f;

__device__ __forceinline__ __bf16 f2bf(float f){
  unsigned u = __builtin_bit_cast(unsigned, f);
  unsigned r = (u + 0x7FFFu + ((u >> 16) & 1u)) >> 16;
  unsigned short s = (unsigned short)r;
  return __builtin_bit_cast(__bf16, s);
}

__device__ __forceinline__ v8f vzero8(){ v8f z = {0.f,0.f,0.f,0.f,0.f,0.f,0.f,0.f}; return z; }

__device__ __forceinline__ v8f wmma_bf16(v16bf a, v16bf b, v8f c){
  return __builtin_amdgcn_wmma_f32_16x16x32_bf16(false, a, false, b, (short)0, c, false, false);
}

// ---- A operand (16x32, row-major [M,K] source) -----------------------------
// lane m=lane&15; contiguous K per lane: k = e + 16*half
__device__ __forceinline__ v16bf fragA_nk(const __bf16* base, int ld, int lane){
  const int half = lane >> 4, m = lane & 15;
  const __bf16* p = base + (size_t)m * ld + 16 * half;
  v16bf f;
#pragma unroll
  for (int e = 0; e < 16; ++e){
    int k = e;
    f[e] = p[k];
  }
  return f;
}

// ---- B operand (32x16) from row-major [N,K] source (weights) ---------------
// B layout used here (same k order as fragA_nk): lane n=lane&15; contiguous K per lane: k = e + 16*half
__device__ __forceinline__ v16bf fragB_nk(const __bf16* base, int ld, int lane){
  const int half = lane >> 4, n = lane & 15;
  const __bf16* p = base + (size_t)n * ld + 16 * half;
  v16bf f;
#pragma unroll
  for (int e = 0; e < 16; ++e) f[e] = p[e];
  return f;
}

// ---- B operand (32x16) from row-major [K,N] source (V tiles, key-major) ----
__device__ __forceinline__ v16bf fragB_kn(const __bf16* base, int ld, int lane){
  const int half = lane >> 4, n = lane & 15;
  const __bf16* p = base + n + (size_t)(16 * half) * ld;
  v16bf f;
#pragma unroll
  for (int e = 0; e < 16; ++e) f[e] = p[(size_t)e * ld];
  return f;
}

// ---- A operand from fp32 LDS tile [16, ld] with on-the-fly bf16 convert ----
__device__ __forceinline__ v16bf fragA_f32(const float* base, int ld, int lane){
  const int half = lane >> 4, m = lane & 15;
  const float* p = base + m * ld + 16 * half;
  v16bf f;
#pragma unroll
  for (int e = 0; e < 16; ++e){
    int k = e;
    f[e] = f2bf(p[k]);
  }
  return f;
}

__device__ __forceinline__ float gelu_exact(float x){
  return 0.5f * x * (1.0f + erff(x * 0.7071067811865476f));
}

// ---------------------------------------------------------------- converts
__global__ void f32_to_bf16_kernel(const float* __restrict__ src,
                                   __bf16* __restrict__ dst, int n){
  int i = blockIdx.x * 256 + threadIdx.x;
  if (i < n) dst[i] = f2bf(src[i]);
}

// -------------------- register-blocked WMMA GEMM: 64x64 tile per wave -------
// C[M,N] = act(A[M,K] @ B[N,K]^T + bias); 4x4 WMMA tiles, 4x fragment reuse.
__global__ __launch_bounds__(32)
void gemm64_bf16_kernel(const __bf16* __restrict__ A, int lda,
                        const __bf16* __restrict__ B, int ldb,
                        const float* __restrict__ bias,
                        float* __restrict__ C, int ldc,
                        __bf16* __restrict__ Cb, int ldcb,
                        int K, int act){
  const int lane = threadIdx.x;
  const int n0 = blockIdx.x * 64;
  const int m0 = blockIdx.y * 64;
  v8f acc[4][4];
#pragma unroll
  for (int i = 0; i < 4; ++i)
#pragma unroll
    for (int j = 0; j < 4; ++j) acc[i][j] = vzero8();

  const __bf16* a_ptr = A + (size_t)m0 * lda;
  const __bf16* b_ptr = B + (size_t)n0 * ldb;
  for (int k0 = 0; k0 < K; k0 += 32){
    // prefetch K tile 8 steps ahead (global_prefetch_b8, speculative)
    __builtin_prefetch(a_ptr + (size_t)(lane & 15) * lda + k0 + 256, 0, 0);
    __builtin_prefetch(b_ptr + (size_t)(lane & 15) * ldb + k0 + 256, 0, 0);
    v16bf af[4], bf_[4];
#pragma unroll
    for (int i = 0; i < 4; ++i)
      af[i]  = fragA_nk(a_ptr + (size_t)(16 * i) * lda + k0, lda, lane);
#pragma unroll
    for (int j = 0; j < 4; ++j)
      bf_[j] = fragB_nk(b_ptr + (size_t)(16 * j) * ldb + k0, ldb, lane);
#pragma unroll
    for (int i = 0; i < 4; ++i)
#pragma unroll
      for (int j = 0; j < 4; ++j)
        acc[i][j] = wmma_bf16(af[i], bf_[j], acc[i][j]);
  }
  const int half = lane >> 4, n = lane & 15;
#pragma unroll
  for (int j = 0; j < 4; ++j){
    const int nc = n0 + 16 * j + n;
    const float bb = bias ? bias[nc] : 0.0f;
#pragma unroll
    for (int i = 0; i < 4; ++i){
#pragma unroll
      for (int r = 0; r < 8; ++r){
        int m = m0 + 16 * i + r + 8 * half;
        float v = acc[i][j][r] + bb;
        if (act == 1) v = gelu_exact(v);
        if (C)  C [(size_t)m * ldc  + nc] = v;
        if (Cb) Cb[(size_t)m * ldcb + nc] = f2bf(v);
      }
    }
  }
}

// ------------------------------------------ split qkv fp32 -> q/k/v bf16
__global__ void extract_qkv_kernel(const float* __restrict__ qkv,
                                   __bf16* __restrict__ qb,
                                   __bf16* __restrict__ kb,
                                   __bf16* __restrict__ vb){
  size_t idx = (size_t)blockIdx.x * 256 + threadIdx.x;
  if (idx >= (size_t)T_ * QKVD_) return;
  int t = (int)(idx / QKVD_);
  int c = (int)(idx % QKVD_);
  __bf16 v = f2bf(qkv[idx]);
  if (c < HD_)              qb[(size_t)t * HD_  + c]          = v;
  else if (c < HD_ + KVD_)  kb[(size_t)t * KVD_ + (c - HD_)]  = v;
  else                      vb[(size_t)t * KVD_ + (c - HD_ - KVD_)] = v;
}

// ---------------------------------- gather compression input blocks (+pos)
__global__ void gather_cmp_kernel(const float* __restrict__ qkv,
                                  const float* __restrict__ cmp_pos,
                                  __bf16* __restrict__ blkK,
                                  __bf16* __restrict__ blkV){
  size_t idx = (size_t)blockIdx.x * 256 + threadIdx.x;
  const size_t total = (size_t)TC_ * KV_ * L_ * D_;
  if (idx >= total) return;
  int row = (int)(idx / (L_ * D_));
  int col = (int)(idx % (L_ * D_));
  int c = row / KV_, g = row % KV_;
  int l = col / D_,  d = col % D_;
  int t = c * S_ + l;
  int off = (blockIdx.z == 0) ? (H_ * D_ + g * D_ + d)
                              : ((H_ + KV_) * D_ + g * D_ + d);
  float v = qkv[(size_t)t * QKVD_ + off] + cmp_pos[l * D_ + d];
  if (blockIdx.z == 0) blkK[idx] = f2bf(v); else blkV[idx] = f2bf(v);
}

// -------------------------- compressed attention + block-score partials
__global__ __launch_bounds__(32)
void cmp_attn_kernel(const __bf16* __restrict__ qb,
                     const __bf16* __restrict__ kcmp,   // [128c][KV][D] bf16
                     const __bf16* __restrict__ vcmp,
                     float* __restrict__ o_cmp,         // [T][H*D]
                     float* __restrict__ blk_part){     // [H][T][NS]
  __shared__ float P[16][128];
  const int lane = threadIdx.x;
  const int t0 = blockIdx.x * 16;
  const int h  = blockIdx.y;
  const int kvh = h / REP_;
  const int half = lane >> 4, nn = lane & 15;

  v16bf qa[4];
  const __bf16* qbase = qb + (size_t)t0 * HD_ + h * D_;
#pragma unroll
  for (int kk = 0; kk < 4; ++kk) qa[kk] = fragA_nk(qbase + 32 * kk, HD_, lane);

#pragma unroll
  for (int ct = 0; ct < 8; ++ct){
    v8f acc = vzero8();
    const __bf16* bbase = kcmp + (size_t)(ct * 16) * 512 + kvh * D_;
#pragma unroll
    for (int kk = 0; kk < 4; ++kk)
      acc = wmma_bf16(qa[kk], fragB_nk(bbase + 32 * kk, 512, lane), acc);
#pragma unroll
    for (int i = 0; i < 8; ++i)
      P[i + 8 * half][ct * 16 + nn] = acc[i] * SCALE_;
  }
  __syncthreads();

  if (lane < 16){
    const int t = t0 + lane;
    int climit = (t >= 31) ? (t - 31) / 16 : -1;          // valid: c*16+31 <= t
    if (climit > TC_ - 1) climit = TC_ - 1;
    if (climit < 0){
      for (int c = 0; c < 128; ++c) P[lane][c] = 0.0f;
    } else {
      float m = -1e30f;
      for (int c = 0; c <= climit; ++c){ float v = P[lane][c]; if (v > m) m = v; }
      float ssum = 0.0f;
      for (int c = 0; c < 128; ++c){
        float v = (c <= climit) ? __expf(P[lane][c] - m) : 0.0f;
        P[lane][c] = v; ssum += v;
      }
      float inv = (ssum > 0.0f) ? 1.0f / ssum : 0.0f;
      for (int c = 0; c < 128; ++c) P[lane][c] *= inv;
    }
    // block score partials: ov overlap -> c in [4j-1, 4j+3]
    for (int j = 0; j < NS_; ++j){
      int clo = 4 * j - 1; if (clo < 0) clo = 0;
      int chi = 4 * j + 3; if (chi > TC_ - 1) chi = TC_ - 1;
      float s = 0.0f;
      for (int c = clo; c <= chi; ++c) s += P[lane][c];
      blk_part[((size_t)h * T_ + t) * NS_ + j] = s;
    }
  }
  __syncthreads();

#pragma unroll
  for (int nt = 0; nt < 8; ++nt){
    v8f acc = vzero8();
#pragma unroll
    for (int kk = 0; kk < 4; ++kk){
      v16bf a = fragA_f32(&P[0][0] + 32 * kk, 128, lane);
      v16bf b = fragB_kn(vcmp + (size_t)(32 * kk) * 512 + kvh * D_ + nt * 16, 512, lane);
      acc = wmma_bf16(a, b, acc);
    }
#pragma unroll
    for (int i = 0; i < 8; ++i)
      o_cmp[(size_t)(t0 + i + 8 * half) * HD_ + h * D_ + nt * 16 + nn] = acc[i];
  }
}

// --------------------------- top-k block selection -> 32-bit allowed mask
__global__ void topk_kernel(const float* __restrict__ blk_part,
                            unsigned* __restrict__ allowed){
  int t = blockIdx.x * 256 + threadIdx.x;
  if (t >= T_) return;
  float s[NS_];
  for (int j = 0; j < NS_; ++j){
    float a = 0.0f;
    for (int h = 0; h < H_; ++h) a += blk_part[((size_t)h * T_ + t) * NS_ + j];
    s[j] = a;
  }
  int cur = t / LP_;
  for (int j = 0; j < NS_; ++j) if (j >= cur) s[j] = -INFINITY;
  unsigned mask = 1u;                       // fixed block 0
  int a2 = cur - 2; if (a2 < 0) a2 = 0; mask |= 1u << a2;
  int a1 = cur - 1; if (a1 < 0) a1 = 0; mask |= 1u << a1;
  mask |= 1u << cur;                        // own block dense
  unsigned used = 0;
  for (int it = 0; it < 13; ++it){          // TOPK-3 dynamic blocks
    int arg = -1; float best = -INFINITY;
    for (int j = 0; j < NS_; ++j)
      if (!((used >> j) & 1u) && s[j] > best){ best = s[j]; arg = j; }
    if (arg < 0){
      for (int j = 0; j < NS_; ++j) if (!((used >> j) & 1u)){ arg = j; break; }
    }
    used |= 1u << arg; mask |= 1u << arg;
  }
  allowed[t] = mask;
}

// ----------------- fused SLC + SWA flash attention (1 wave, 16 q rows)
__global__ __launch_bounds__(32)
void slc_swa_kernel(const __bf16* __restrict__ qb,
                    const __bf16* __restrict__ kb,
                    const __bf16* __restrict__ vb,
                    const unsigned* __restrict__ allowed,
                    const float* __restrict__ sinks,
                    float* __restrict__ o_slc,
                    float* __restrict__ o_swa){
  __shared__ float Lt[16][32], Ps[16][32], Pw[16][32];
  __shared__ float fac_s[16], fac_w[16], inv_s[16], inv_w[16];
  const int lane = threadIdx.x;
  const int t0 = blockIdx.x * 16;
  const int h  = blockIdx.y;
  const int kvh = h / REP_;
  const int half = lane >> 4, nn = lane & 15;

  v16bf qa[4];
  const __bf16* qbase = qb + (size_t)t0 * HD_ + h * D_;
#pragma unroll
  for (int kk = 0; kk < 4; ++kk) qa[kk] = fragA_nk(qbase + 32 * kk, HD_, lane);

  v8f os[8], ow[8];
#pragma unroll
  for (int nt = 0; nt < 8; ++nt){ os[nt] = vzero8(); ow[nt] = vzero8(); }

  float m_s = -1e30f, l_s = 0.0f;
  float m_w = sinks[h], l_w = 1.0f;   // sink folded into SWA online softmax

  const int nchunk = (t0 + 16 + 31) >> 5;
  for (int sc = 0; sc < nchunk; ++sc){
    const int s0 = sc * 32;
    // ---- logits 16x32 (2 n-tiles x 4 k-steps of WMMA) ----
#pragma unroll
    for (int nt2 = 0; nt2 < 2; ++nt2){
      v8f acc = vzero8();
      const __bf16* bbase = kb + (size_t)(s0 + nt2 * 16) * KVD_ + kvh * D_;
#pragma unroll
      for (int kk = 0; kk < 4; ++kk)
        acc = wmma_bf16(qa[kk], fragB_nk(bbase + 32 * kk, KVD_, lane), acc);
#pragma unroll
      for (int i = 0; i < 8; ++i)
        Lt[i + 8 * half][nt2 * 16 + nn] = acc[i] * SCALE_;
    }
    __syncthreads();
    // ---- dual masked online softmax (rows on lanes 0..15) ----
    if (lane < 16){
      const int t = t0 + lane;
      const unsigned am = allowed[t];
      float nm_s = m_s, nm_w = m_w;
      for (int j = 0; j < 32; ++j){
        int s = s0 + j; if (s > t) break;
        float v = Lt[lane][j];
        if (((am >> ((unsigned)s >> 6)) & 1u) && v > nm_s) nm_s = v;
        if ((s > t - WIN_) && v > nm_w) nm_w = v;
      }
      float f_s = __expf(m_s - nm_s), f_w = __expf(m_w - nm_w);
      float sum_s = 0.0f, sum_w = 0.0f;
      for (int j = 0; j < 32; ++j){
        int s = s0 + j;
        bool causal = (s <= t);
        float v = causal ? Lt[lane][j] : 0.0f;
        bool ok_s = causal && (((am >> ((unsigned)s >> 6)) & 1u) != 0u);
        bool ok_w = causal && (s > t - WIN_);
        float es = ok_s ? __expf(v - nm_s) : 0.0f;
        float ew = ok_w ? __expf(v - nm_w) : 0.0f;
        Ps[lane][j] = es; Pw[lane][j] = ew;
        sum_s += es; sum_w += ew;
      }
      l_s = l_s * f_s + sum_s;  l_w = l_w * f_w + sum_w;
      m_s = nm_s;               m_w = nm_w;
      fac_s[lane] = f_s;        fac_w[lane] = f_w;
    }
    __syncthreads();
    // ---- rescale accumulators, then PV WMMA (K=32) ----
    v16bf aS = fragA_f32(&Ps[0][0], 32, lane);
    v16bf aW = fragA_f32(&Pw[0][0], 32, lane);
#pragma unroll
    for (int nt = 0; nt < 8; ++nt){
      v16bf bV = fragB_kn(vb + (size_t)s0 * KVD_ + kvh * D_ + nt * 16, KVD_, lane);
#pragma unroll
      for (int i = 0; i < 8; ++i){
        os[nt][i] *= fac_s[i + 8 * half];
        ow[nt][i] *= fac_w[i + 8 * half];
      }
      os[nt] = wmma_bf16(aS, bV, os[nt]);
      ow[nt] = wmma_bf16(aW, bV, ow[nt]);
    }
    __syncthreads();
  }
  if (lane < 16){
    inv_s[lane] = (l_s > 0.0f) ? 1.0f / l_s : 0.0f;
    inv_w[lane] = (l_w > 0.0f) ? 1.0f / l_w : 0.0f;
  }
  __syncthreads();
#pragma unroll
  for (int nt = 0; nt < 8; ++nt){
#pragma unroll
    for (int i = 0; i < 8; ++i){
      size_t off = (size_t)(t0 + i + 8 * half) * HD_ + h * D_ + nt * 16 + nn;
      o_slc[off] = os[nt][i] * inv_s[i + 8 * half];
      o_swa[off] = ow[nt][i] * inv_w[i + 8 * half];
    }
  }
}

// ---------------------------------------- gate layer 2 (N=3) + sigmoid
__global__ void gate2_kernel(const float* __restrict__ hid,
                             const float* __restrict__ Wg2,
                             const float* __restrict__ bg2,
                             float* __restrict__ gates){
  int t = blockIdx.x * 256 + threadIdx.x;
  if (t >= T_) return;
  for (int e = 0; e < 3; ++e){
    float a = bg2[e];
    const float* hrow = hid + (size_t)t * GH_;
    const float* wrow = Wg2 + (size_t)e * GH_;
    for (int k = 0; k < GH_; ++k) a += hrow[k] * wrow[k];
    gates[t * 3 + e] = 1.0f / (1.0f + __expf(-a));
  }
}

// ---------------------------------------- gated combine -> bf16
__global__ void combine_kernel(const float* __restrict__ o_cmp,
                               const float* __restrict__ o_slc,
                               const float* __restrict__ o_swa,
                               const float* __restrict__ gates,
                               __bf16* __restrict__ ob){
  size_t idx = (size_t)blockIdx.x * 256 + threadIdx.x;
  if (idx >= (size_t)T_ * HD_) return;
  int t = (int)(idx / HD_);
  float g0 = gates[t * 3 + 0], g1 = gates[t * 3 + 1], g2 = gates[t * 3 + 2];
  ob[idx] = f2bf(g0 * o_cmp[idx] + g1 * o_slc[idx] + g2 * o_swa[idx]);
}

// =======================================================================
extern "C" void kernel_launch(void* const* d_in, const int* in_sizes, int n_in,
                              void* d_out, int out_size, void* d_ws, size_t ws_size,
                              hipStream_t stream) {
  const float* x      = (const float*)d_in[0];
  const float* W_qkv  = (const float*)d_in[1];
  const float* b_qkv  = (const float*)d_in[2];
  const float* W_out  = (const float*)d_in[3];
  const float* b_out  = (const float*)d_in[4];
  const float* sinks  = (const float*)d_in[5];
  const float* cmp_pos= (const float*)d_in[6];
  const float* W_c1   = (const float*)d_in[7];
  const float* b_c1   = (const float*)d_in[8];
  const float* W_c2   = (const float*)d_in[9];
  const float* b_c2   = (const float*)d_in[10];
  const float* W_g1   = (const float*)d_in[11];
  const float* b_g1   = (const float*)d_in[12];
  const float* W_g2   = (const float*)d_in[13];
  const float* b_g2   = (const float*)d_in[14];
  float* out = (float*)d_out;

  char* wp = (char*)d_ws;
  auto alloc = [&](size_t bytes)->char*{
    char* r = wp; wp += (bytes + 255) & ~(size_t)255; return r;
  };
  __bf16* xb     = (__bf16*)alloc((size_t)T_ * DIM_ * 2);
  __bf16* Wqkv_b = (__bf16*)alloc((size_t)QKVD_ * DIM_ * 2);
  __bf16* Wout_b = (__bf16*)alloc((size_t)DIM_ * HD_ * 2);
  __bf16* Wg1_b  = (__bf16*)alloc((size_t)GH_ * DIM_ * 2);
  __bf16* Wc1_b  = (__bf16*)alloc((size_t)CMPH_ * (L_ * D_) * 2);
  __bf16* Wc2_b  = (__bf16*)alloc((size_t)D_ * CMPH_ * 2);
  float*  qkv    = (float*) alloc((size_t)T_ * QKVD_ * 4);
  __bf16* qb     = (__bf16*)alloc((size_t)T_ * HD_ * 2);
  __bf16* kb     = (__bf16*)alloc((size_t)T_ * KVD_ * 2);
  __bf16* vb     = (__bf16*)alloc((size_t)T_ * KVD_ * 2);
  __bf16* blkK   = (__bf16*)alloc((size_t)512 * (L_ * D_) * 2);   // 508 rows used
  __bf16* blkV   = (__bf16*)alloc((size_t)512 * (L_ * D_) * 2);
  __bf16* hidKb  = (__bf16*)alloc((size_t)512 * CMPH_ * 2);
  __bf16* hidVb  = (__bf16*)alloc((size_t)512 * CMPH_ * 2);
  __bf16* kcmp_b = (__bf16*)alloc((size_t)512 * D_ * 2);          // [128c][4g][128d]
  __bf16* vcmp_b = (__bf16*)alloc((size_t)512 * D_ * 2);
  float*  blk_pt = (float*) alloc((size_t)H_ * T_ * NS_ * 4);
  unsigned* amask= (unsigned*)alloc((size_t)T_ * 4);
  float*  o_cmp  = (float*) alloc((size_t)T_ * HD_ * 4);
  float*  o_slc  = (float*) alloc((size_t)T_ * HD_ * 4);
  float*  o_swa  = (float*) alloc((size_t)T_ * HD_ * 4);
  float*  hid_g  = (float*) alloc((size_t)T_ * GH_ * 4);
  float*  gates  = (float*) alloc((size_t)T_ * 3 * 4);
  __bf16* ob     = (__bf16*)alloc((size_t)T_ * HD_ * 2);

  auto cvt = [&](const float* s, __bf16* d, int n){
    f32_to_bf16_kernel<<<(n + 255) / 256, 256, 0, stream>>>(s, d, n);
  };
  cvt(x,     xb,     T_ * DIM_);
  cvt(W_qkv, Wqkv_b, QKVD_ * DIM_);
  cvt(W_out, Wout_b, DIM_ * HD_);
  cvt(W_g1,  Wg1_b,  GH_ * DIM_);
  cvt(W_c1,  Wc1_b,  CMPH_ * L_ * D_);
  cvt(W_c2,  Wc2_b,  D_ * CMPH_);

  // QKV projection: [T,3072] = x @ Wqkv^T + b
  gemm64_bf16_kernel<<<dim3(QKVD_ / 64, T_ / 64), 32, 0, stream>>>(
      xb, DIM_, Wqkv_b, DIM_, b_qkv, qkv, QKVD_, nullptr, 0, DIM_, 0);

  extract_qkv_kernel<<<(T_ * QKVD_ + 255) / 256, 256, 0, stream>>>(qkv, qb, kb, vb);

  // compression: gather blocks, MLP (gelu) -> k_cmp / v_cmp
  {
    int total = TC_ * KV_ * L_ * D_;
    gather_cmp_kernel<<<dim3((total + 255) / 256, 1, 2), 256, 0, stream>>>(
        qkv, cmp_pos, blkK, blkV);
  }
  gemm64_bf16_kernel<<<dim3(CMPH_ / 64, 512 / 64), 32, 0, stream>>>(
      blkK, L_ * D_, Wc1_b, L_ * D_, b_c1, nullptr, 0, hidKb, CMPH_, L_ * D_, 1);
  gemm64_bf16_kernel<<<dim3(CMPH_ / 64, 512 / 64), 32, 0, stream>>>(
      blkV, L_ * D_, Wc1_b, L_ * D_, b_c1, nullptr, 0, hidVb, CMPH_, L_ * D_, 1);
  gemm64_bf16_kernel<<<dim3(D_ / 64, 512 / 64), 32, 0, stream>>>(
      hidKb, CMPH_, Wc2_b, CMPH_, b_c2, nullptr, 0, kcmp_b, D_, CMPH_, 0);
  gemm64_bf16_kernel<<<dim3(D_ / 64, 512 / 64), 32, 0, stream>>>(
      hidVb, CMPH_, Wc2_b, CMPH_, b_c2, nullptr, 0, vcmp_b, D_, CMPH_, 0);

  // compressed attention + block score partials
  cmp_attn_kernel<<<dim3(T_ / 16, H_), 32, 0, stream>>>(
      qb, kcmp_b, vcmp_b, o_cmp, blk_pt);

  // top-k block selection -> allowed bitmask
  topk_kernel<<<(T_ + 255) / 256, 256, 0, stream>>>(blk_pt, amask);

  // fused SLC + SWA flash attention
  slc_swa_kernel<<<dim3(T_ / 16, H_), 32, 0, stream>>>(
      qb, kb, vb, amask, sinks, o_slc, o_swa);

  // gate MLP
  gemm64_bf16_kernel<<<dim3(GH_ / 64, T_ / 64), 32, 0, stream>>>(
      xb, DIM_, Wg1_b, DIM_, b_g1, hid_g, GH_, nullptr, 0, DIM_, 1);
  gate2_kernel<<<(T_ + 255) / 256, 256, 0, stream>>>(hid_g, W_g2, b_g2, gates);

  // gated combine + output projection -> d_out
  combine_kernel<<<(T_ * HD_ + 255) / 256, 256, 0, stream>>>(
      o_cmp, o_slc, o_swa, gates, ob);
  gemm64_bf16_kernel<<<dim3(DIM_ / 64, T_ / 64), 32, 0, stream>>>(
      ob, HD_, Wout_b, HD_, b_out, out, DIM_, nullptr, 0, HD_, 0);
}
